// MultiHeadAttention_83365315215983
// MI455X (gfx1250) — matrix operand order repaired and checked by reading, not compiled, not run on hardware
//
#include <hip/hip_runtime.h>
#include <hip/hip_bf16.h>

// ---------------------------------------------------------------------------
// MHA forward for MI455X (gfx1250, wave32, WMMA).  B=4,S=2048,E=1024,H=16,D=64
// Pipeline (all launches serialized on `stream`):
//   per projection: cvt_bf16(X) once + wtrans(W->bf16 N-major) once, then a
//   pure-WMMA GEMM with contiguous global fragment loads (no LDS, no barriers)
//   -> flash-attention kernel (online softmax, WMMA QK^T and P@V, LDS P-stage)
//   -> output projection to f32.
// ---------------------------------------------------------------------------

typedef __attribute__((ext_vector_type(16))) __bf16 v16bf;
typedef __attribute__((ext_vector_type(8)))  __bf16 v8bf;
typedef __attribute__((ext_vector_type(8)))  float  v8f;

#define S_LEN 2048
#define D_HEAD 64
#define NHEAD 16
#define EMB 1024

static __device__ __forceinline__ v8f wmma_bf16(v16bf a, v16bf b, v8f c) {
  return __builtin_amdgcn_wmma_f32_16x16x32_bf16(
      /*neg_a=*/false, a, /*neg_b=*/false, b,
      /*c_mod=*/(short)0, c, /*reuse_a=*/false, /*reuse_b=*/false);
}

// Build a 16x32 bf16 A-fragment for one lane: one contiguous 16-element run
// at (base) (contiguous k order, matching the B loads).
static __device__ __forceinline__ v16bf load_afrag(const __bf16* p) {
  v8bf r1 = *(const v8bf*)(p);
  v8bf r2 = *(const v8bf*)(p + 8);
  return __builtin_shufflevector(r1, r2, 0, 1, 2, 3, 4, 5, 6, 7,
                                          8, 9, 10, 11, 12, 13, 14, 15);
}

// ---------------------------------------------------------------------------
// Elementwise f32 -> bf16 convert (vectorized float4 in, 4x bf16 out).
// ---------------------------------------------------------------------------
typedef __attribute__((ext_vector_type(4))) __bf16 v4bf;

__global__ __launch_bounds__(256) void cvt_bf16_kernel(
    const float* __restrict__ in, __bf16* __restrict__ out, int n4)
{
  int i = blockIdx.x * 256 + threadIdx.x;
  if (i < n4) {
    float4 v = ((const float4*)in)[i];
    v4bf o = {(__bf16)v.x, (__bf16)v.y, (__bf16)v.z, (__bf16)v.w};
    ((v4bf*)out)[i] = o;
  }
}

// ---------------------------------------------------------------------------
// W (K x N, f32) -> Wt (N x K, bf16).  32x32 LDS tile, coalesced both ways.
// ---------------------------------------------------------------------------
__global__ __launch_bounds__(256) void wtrans_kernel(
    const float* __restrict__ W, __bf16* __restrict__ Wt)
{
  __shared__ float t[32][33];
  const int nb = blockIdx.x * 32;
  const int kb = blockIdx.y * 32;
  const int tx = threadIdx.x & 31;
  const int ty = threadIdx.x >> 5;       // 0..7
  #pragma unroll
  for (int j = 0; j < 4; ++j) {
    int k = ty + j * 8;
    t[k][tx] = W[(size_t)(kb + k) * EMB + nb + tx];
  }
  __syncthreads();
  #pragma unroll
  for (int j = 0; j < 4; ++j) {
    int n = ty + j * 8;
    Wt[(size_t)(nb + n) * EMB + kb + tx] = (__bf16)t[tx][n];
  }
}

// ---------------------------------------------------------------------------
// GEMM: Y[M=8192, N=1024] = Xb[M,K] @ Wt^T + bias  (Xb bf16, Wt bf16 N-major)
// Block = 128 threads (4 waves). Wave tile = 32 rows x 64 cols
//   -> 8 C-fragments, inner loop = 8 WMMAs per 32-k chunk, all fragment
//      loads are contiguous 16B/32B global loads (Wt lives in L2: 2MB).
// mode 0: bf16 (BH,S,D); mode 1: bf16 (BH,D,S) [V^T]; mode 2: f32 (M,E)
// ---------------------------------------------------------------------------
__global__ __launch_bounds__(128) void proj_kernel(
    const __bf16* __restrict__ xb,
    const __bf16* __restrict__ Wt,
    const float* __restrict__ bias,
    float* __restrict__ outf,
    __bf16* __restrict__ outb,
    int mode, float out_scale)
{
  const int lane = threadIdx.x & 31;
  const int wave = threadIdx.x >> 5;
  const int half = lane >> 4;
  const int l16  = lane & 15;

  const int nb = blockIdx.x * 64;
  const int mb = blockIdx.y * 128 + wave * 32;

  const __bf16* x0 = xb + (size_t)(mb + l16) * EMB + half * 16;
  const __bf16* x1 = xb + (size_t)(mb + 16 + l16) * EMB + half * 16;

  v8f acc[8];
  #pragma unroll
  for (int i = 0; i < 8; ++i) acc[i] = (v8f){};

  for (int kc = 0; kc < EMB; kc += 32) {
    v16bf a0 = load_afrag(x0 + kc);
    v16bf a1 = load_afrag(x1 + kc);
    #pragma unroll
    for (int nt = 0; nt < 4; ++nt) {
      const __bf16* wp =
          Wt + (size_t)(nb + nt * 16 + l16) * EMB + kc + half * 16;
      v16bf bfrag = *(const v16bf*)wp;       // contiguous 32B, 32B-aligned
      acc[nt]     = wmma_bf16(a0, bfrag, acc[nt]);
      acc[4 + nt] = wmma_bf16(a1, bfrag, acc[4 + nt]);
    }
  }

  // Epilogue: C layout -> element r is row (r + half*8), column l16.
  #pragma unroll
  for (int mt = 0; mt < 2; ++mt) {
    #pragma unroll
    for (int nt = 0; nt < 4; ++nt) {
      const int c = nb + nt * 16 + l16;
      const float bv = bias[c];
      v8f a = acc[mt * 4 + nt];
      #pragma unroll
      for (int r = 0; r < 8; ++r) {
        const int g = mb + mt * 16 + r + half * 8;
        const float val = (a[r] + bv) * out_scale;
        if (mode == 2) {
          outf[(size_t)g * EMB + c] = val;
        } else {
          const int bidx = g >> 11;            // g / S_LEN
          const int s    = g & (S_LEN - 1);
          const int h    = c >> 6;
          const int d    = c & (D_HEAD - 1);
          if (mode == 0)
            outb[(((size_t)bidx * NHEAD + h) * S_LEN + s) * D_HEAD + d] =
                (__bf16)val;
          else
            outb[(((size_t)bidx * NHEAD + h) * D_HEAD + d) * S_LEN + s] =
                (__bf16)val;
        }
      }
    }
  }
}

// ---------------------------------------------------------------------------
// Flash attention: 8 waves/block, each wave owns one 16-row Q tile of one
// (b,h). Keys streamed 32 at a time: 4 WMMAs for scores, per-lane online
// softmax (C layout = 1 column x 8 rows per lane, 16-wide shfl_xor row
// reductions), P re-laid out C->A through a 1KB/wave LDS stage, 4 WMMAs P@V.
// ---------------------------------------------------------------------------
__global__ __launch_bounds__(256) void attn_kernel(
    const __bf16* __restrict__ qb,   // (BH,S,D), pre-scaled by 1/sqrt(D)
    const __bf16* __restrict__ kb,   // (BH,S,D)
    const __bf16* __restrict__ vt,   // (BH,D,S) transposed
    const unsigned char* __restrict__ mask, // (B,1,1,S) bool
    __bf16* __restrict__ attnb)      // (B,S,E)
{
  __shared__ __attribute__((aligned(64))) __bf16 pstage[8][16 * 32];

  const int lane = threadIdx.x & 31;
  const int wave = threadIdx.x >> 5;
  const int half = lane >> 4;
  const int l16  = lane & 15;

  const int bh = blockIdx.y;         // 0..63
  const int b  = bh >> 4;
  const int h  = bh & 15;
  const int qbase = (blockIdx.x * 8 + wave) * 16;

  // Q A-fragments for K-dim chunks [0,32) and [32,64).
  const __bf16* qrow = qb + ((size_t)bh * S_LEN + qbase + l16) * D_HEAD;
  v16bf qa0 = load_afrag(qrow + half * 16);
  v16bf qa1 = load_afrag(qrow + 32 + half * 16);

  v8f o0 = {}, o1 = {}, o2 = {}, o3 = {};
  float m[8], l[8];
  #pragma unroll
  for (int r = 0; r < 8; ++r) { m[r] = -3.0e38f; l[r] = 0.0f; }

  __bf16* ps = &pstage[wave][0];

  for (int kt = 0; kt < S_LEN; kt += 32) {
    if (kt + 32 < S_LEN)
      __builtin_prefetch(kb + ((size_t)bh * S_LEN + kt + 32 + l16) * D_HEAD,
                         0, 1);

    // ---- scores: two 16-key tiles, accumulating over d in 2 chunks ----
    v8f s0 = {}, s1 = {};
    {
      const __bf16* kr0 =
          kb + ((size_t)bh * S_LEN + kt + l16) * D_HEAD + half * 16;
      const __bf16* kr1 = kr0 + 16 * D_HEAD;   // keys kt+16..kt+31
      v16bf kb00 = *(const v16bf*)(kr0);        // d 0..31
      v16bf kb10 = *(const v16bf*)(kr1);
      v16bf kb01 = *(const v16bf*)(kr0 + 32);   // d 32..63
      v16bf kb11 = *(const v16bf*)(kr1 + 32);
      s0 = wmma_bf16(qa0, kb00, s0);
      s1 = wmma_bf16(qa0, kb10, s1);
      s0 = wmma_bf16(qa1, kb01, s0);
      s1 = wmma_bf16(qa1, kb11, s1);
    }

    // ---- mask + online softmax (per-lane column, 8 rows) ----
    const bool ok0 = mask[(size_t)b * S_LEN + kt + l16] != 0;
    const bool ok1 = mask[(size_t)b * S_LEN + kt + 16 + l16] != 0;

    float p0[8], p1[8];
    #pragma unroll
    for (int r = 0; r < 8; ++r) {
      const float a0v = ok0 ? s0[r] : -3.0e38f;
      const float a1v = ok1 ? s1[r] : -3.0e38f;
      float t = fmaxf(a0v, a1v);
      #pragma unroll
      for (int off = 1; off < 16; off <<= 1)
        t = fmaxf(t, __shfl_xor(t, off, 16));   // row-max across 16 lanes
      const float mnew = fmaxf(m[r], t);
      p0[r] = ok0 ? __expf(s0[r] - mnew) : 0.0f;
      p1[r] = ok1 ? __expf(s1[r] - mnew) : 0.0f;
      float rs = p0[r] + p1[r];
      #pragma unroll
      for (int off = 1; off < 16; off <<= 1)
        rs += __shfl_xor(rs, off, 16);          // row-sum across 16 lanes
      const float corr = __expf(m[r] - mnew);
      l[r] = l[r] * corr + rs;
      o0[r] *= corr; o1[r] *= corr; o2[r] *= corr; o3[r] *= corr;
      m[r] = mnew;
    }

    // ---- re-layout P (C-frag -> A-frag) through LDS ----
    #pragma unroll
    for (int r = 0; r < 8; ++r) {
      const int prow = r + half * 8;
      ps[prow * 32 + l16]      = (__bf16)p0[r];
      ps[prow * 32 + 16 + l16] = (__bf16)p1[r];
    }
    v16bf pa = load_afrag(ps + l16 * 32 + half * 16);

    // ---- P(16x32) @ V(32x64): 4 d-tiles, contiguous 32B loads of V^T ----
    const __bf16* vb =
        vt + ((size_t)bh * D_HEAD + l16) * S_LEN + kt + half * 16;
    v16bf vb0 = *(const v16bf*)(vb + 0 * 16 * S_LEN);
    v16bf vb1 = *(const v16bf*)(vb + 1 * 16 * S_LEN);
    v16bf vb2 = *(const v16bf*)(vb + 2 * 16 * S_LEN);
    v16bf vb3 = *(const v16bf*)(vb + 3 * 16 * S_LEN);
    o0 = wmma_bf16(pa, vb0, o0);
    o1 = wmma_bf16(pa, vb1, o1);
    o2 = wmma_bf16(pa, vb2, o2);
    o3 = wmma_bf16(pa, vb3, o3);
  }

  // ---- epilogue: divide by l, scatter to (B,S,E) bf16 ----
  #pragma unroll
  for (int r = 0; r < 8; ++r) {
    const float inv = 1.0f / l[r];
    const int q = qbase + r + half * 8;
    __bf16* orow = attnb + ((size_t)b * S_LEN + q) * EMB + h * D_HEAD;
    orow[0 * 16 + l16] = (__bf16)(o0[r] * inv);
    orow[1 * 16 + l16] = (__bf16)(o1[r] * inv);
    orow[2 * 16 + l16] = (__bf16)(o2[r] * inv);
    orow[3 * 16 + l16] = (__bf16)(o3[r] * inv);
  }
}

// ---------------------------------------------------------------------------
extern "C" void kernel_launch(void* const* d_in, const int* in_sizes, int n_in,
                              void* d_out, int out_size, void* d_ws,
                              size_t ws_size, hipStream_t stream) {
  (void)in_sizes; (void)n_in; (void)out_size; (void)ws_size;
  const float* q  = (const float*)d_in[0];
  const float* k  = (const float*)d_in[1];
  const float* v  = (const float*)d_in[2];
  const unsigned char* mask = (const unsigned char*)d_in[3]; // jnp bool = 1B
  const float* Wq = (const float*)d_in[4];
  const float* bq = (const float*)d_in[5];
  const float* Wk = (const float*)d_in[6];
  const float* bk = (const float*)d_in[7];
  const float* Wv = (const float*)d_in[8];
  const float* bv = (const float*)d_in[9];
  const float* Wo = (const float*)d_in[10];
  const float* bo = (const float*)d_in[11];

  // Workspace layout (buffers reused across the serialized launches):
  //   [0,16M)  qbuf   bf16 (BH,S,D)
  //   [16,32M) kbuf   bf16 (BH,S,D)
  //   [32,48M) vtbuf  bf16 (BH,D,S)
  //   [48,64M) xcvt   bf16 converted input / later attn output (B,S,E)
  //   [64,66M) wtbuf  bf16 transposed weight (N,K)
  const size_t NELT = (size_t)4 * NHEAD * S_LEN * D_HEAD;  // 8388608
  __bf16* qbuf  = (__bf16*)d_ws;
  __bf16* kbuf  = qbuf + NELT;
  __bf16* vtbuf = kbuf + NELT;
  __bf16* xcvt  = vtbuf + NELT;          // doubles as attn output buffer
  __bf16* wtbuf = xcvt + NELT;

  const int n4 = (int)(NELT / 4);
  const dim3 cg((n4 + 255) / 256), cb(256);
  const dim3 tg(EMB / 32, EMB / 32), tb(256);
  const dim3 pg(EMB / 64, (4 * S_LEN) / 128), pb(128);   // 16 x 64

  // Q projection (1/sqrt(D) folded into out_scale)
  cvt_bf16_kernel<<<cg, cb, 0, stream>>>(q, xcvt, n4);
  wtrans_kernel<<<tg, tb, 0, stream>>>(Wq, wtbuf);
  proj_kernel<<<pg, pb, 0, stream>>>(xcvt, wtbuf, bq, nullptr, qbuf, 0, 0.125f);
  // K projection
  cvt_bf16_kernel<<<cg, cb, 0, stream>>>(k, xcvt, n4);
  wtrans_kernel<<<tg, tb, 0, stream>>>(Wk, wtbuf);
  proj_kernel<<<pg, pb, 0, stream>>>(xcvt, wtbuf, bk, nullptr, kbuf, 0, 1.0f);
  // V projection (transposed output)
  cvt_bf16_kernel<<<cg, cb, 0, stream>>>(v, xcvt, n4);
  wtrans_kernel<<<tg, tb, 0, stream>>>(Wv, wtbuf);
  proj_kernel<<<pg, pb, 0, stream>>>(xcvt, wtbuf, bv, nullptr, vtbuf, 1, 1.0f);

  // Attention (writes bf16 (B,S,E) into xcvt, now dead as an input buffer)
  const dim3 ag(S_LEN / (16 * 8), 4 * NHEAD), ab(256);   // 16 x 64
  attn_kernel<<<ag, ab, 0, stream>>>(qbuf, kbuf, vtbuf, mask, xcvt);

  // Output projection -> f32 d_out
  wtrans_kernel<<<tg, tb, 0, stream>>>(Wo, wtbuf);
  proj_kernel<<<pg, pb, 0, stream>>>(xcvt, wtbuf, bo, (float*)d_out, nullptr,
                                     2, 1.0f);
}
